// GroupedQueryAttention_16758962389057
// MI455X (gfx1250) — matrix operand order repaired and checked by reading, not compiled, not run on hardware
//
#include <hip/hip_runtime.h>
#include <hip/hip_bf16.h>

typedef __attribute__((ext_vector_type(16))) __bf16 v16bf;
typedef __attribute__((ext_vector_type(8)))  float  v8f;
typedef __attribute__((ext_vector_type(4)))  int    v4i;

struct BF8 { __bf16 d[8]; };   // 16-byte chunk

__device__ inline void set8(v16bf& f, int off, const __bf16* p) {
  BF8 c = *reinterpret_cast<const BF8*>(p);
#pragma unroll
  for (int i = 0; i < 8; ++i) f[off + i] = c.d[i];
}

// ---------------------------------------------------------------------------
// CDNA5 async global->LDS copy (ASYNCcnt path), builtin if present, else asm
// ---------------------------------------------------------------------------
#if __has_builtin(__builtin_amdgcn_global_load_async_to_lds_b128)
#define ASYNC_VIA_BUILTIN 1
#endif

typedef __attribute__((address_space(1))) v4i* gv4i_p;  // global int4*
typedef __attribute__((address_space(3))) v4i* lv4i_p;  // LDS int4*

__device__ inline void async_ld_b128(const __bf16* g, __bf16* l) {
#ifdef ASYNC_VIA_BUILTIN
  __builtin_amdgcn_global_load_async_to_lds_b128(
      (gv4i_p)(void*)g, (lv4i_p)(void*)l, 0, 0);
#else
  unsigned loff = (unsigned)(unsigned long long)l;  // LDS aperture: low 32 bits = LDS byte addr
  asm volatile("global_load_async_to_lds_b128 %0, %1, off"
               :: "v"(loff), "v"(g) : "memory");
#endif
}

__device__ inline void wait_async_le2() {
#if __has_builtin(__builtin_amdgcn_s_wait_asynccnt)
  __builtin_amdgcn_s_wait_asynccnt(2);
#else
  asm volatile("s_wait_asynccnt 0x2" ::: "memory");
#endif
}
__device__ inline void wait_async_le0() {
#if __has_builtin(__builtin_amdgcn_s_wait_asynccnt)
  __builtin_amdgcn_s_wait_asynccnt(0);
#else
  asm volatile("s_wait_asynccnt 0x0" ::: "memory");
#endif
}

// ---------------------------------------------------------------------------
// fp32 -> bf16 elementwise
// ---------------------------------------------------------------------------
__global__ __launch_bounds__(256)
void cvt_f32_bf16(const float* __restrict__ in, __bf16* __restrict__ out, long long n) {
  long long i = (long long)blockIdx.x * blockDim.x + threadIdx.x;
  long long stride = (long long)gridDim.x * blockDim.x;
  for (; i < n; i += stride) out[i] = (__bf16)in[i];
}

// ---------------------------------------------------------------------------
// Tiled WMMA GEMM: C[M x N] (f32) = A[M x K] (bf16 rowmajor) * B[K x N] (bf16 rowmajor)
// block tile 128x128, 8 waves (4x2), wave tile 32x64, K-step 32.
// Double-buffered LDS; A tile via async-to-LDS, B tile software-pipelined in VGPRs.
// ---------------------------------------------------------------------------
#define GTM 128
#define GTN 128
#define GTK 32
#define LDA 40   // padded bf16 stride (80B: 16B aligned, conflict-free)
#define LDB 40

__global__ __launch_bounds__(256)
void gemm_bf16(const __bf16* __restrict__ A, const __bf16* __restrict__ B,
               float* __restrict__ C, int M, int N, int K) {
  __shared__ __bf16 As[2][GTM * LDA];
  __shared__ __bf16 Bt[2][GTN * LDB];   // transposed tile: Bt[n][k]

  const int tid  = threadIdx.x;
  const int lane = tid & 31;
  const int w    = tid >> 5;
  const int hi   = lane >> 4;
  const int l15  = lane & 15;
  const int wm   = (w >> 1) * 32;   // 4 wave-rows
  const int wn   = (w & 1) * 64;    // 2 wave-cols
  const int bm   = blockIdx.y * GTM;
  const int bn   = blockIdx.x * GTN;

  const int arow  = tid >> 1;          // 0..127
  const int acol  = (tid & 1) * 16;    // 0 | 16
  const int brow  = tid >> 3;          // 0..31  (k)
  const int bcolc = (tid & 7) * 16;    // 0..112 (n), 16 cols per thread

  const __bf16* gA = A + (size_t)(bm + arow) * K + acol;
  const __bf16* gB = B + (size_t)brow * N + bn + bcolc;

  const int nk = K / GTK;
  v8f acc[2][4] = {};

  // ---- prologue: async A tile 0, global B tile 0 into VGPRs
  async_ld_b128(gA,     &As[0][arow * LDA + acol]);
  async_ld_b128(gA + 8, &As[0][arow * LDA + acol + 8]);
  BF8 breg0 = *reinterpret_cast<const BF8*>(gB);
  BF8 breg1 = *reinterpret_cast<const BF8*>(gB + 8);

  for (int kt = 0; kt < nk; ++kt) {
    const int  cb   = kt & 1;
    const bool more = (kt + 1 < nk);

    if (more) {  // issue async A for next tile into the other buffer
      const __bf16* ga = gA + (size_t)(kt + 1) * GTK;
      async_ld_b128(ga,     &As[cb ^ 1][arow * LDA + acol]);
      async_ld_b128(ga + 8, &As[cb ^ 1][arow * LDA + acol + 8]);
    }

    // store current B tile transposed into LDS
#pragma unroll
    for (int i = 0; i < 8; ++i) Bt[cb][(bcolc + i) * LDB + brow]     = breg0.d[i];
#pragma unroll
    for (int i = 0; i < 8; ++i) Bt[cb][(bcolc + 8 + i) * LDB + brow] = breg1.d[i];

    if (more) {  // global loads for next B tile; prefetch tile k+2
      const __bf16* nb = gB + (size_t)(kt + 1) * GTK * N;
      breg0 = *reinterpret_cast<const BF8*>(nb);
      breg1 = *reinterpret_cast<const BF8*>(nb + 8);
      if (kt + 2 < nk) {
        __builtin_prefetch(gA + (size_t)(kt + 2) * GTK, 0, 3);
        __builtin_prefetch(gB + (size_t)(kt + 2) * GTK * N, 0, 3);
      }
    }

    // retire the async copy of the CURRENT tile (in-order: <=2 leaves next in flight)
    if (more) wait_async_le2(); else wait_async_le0();
    __syncthreads();

    v16bf af[2], bfv[4];
#pragma unroll
    for (int mi = 0; mi < 2; ++mi) {
      // A 16x32 bf16 layout: lane row = l15; K chunks [hi*8,hi*8+8) and [hi*8+16,hi*8+24)
      const __bf16* pr = &As[cb][(wm + mi * 16 + l15) * LDA + hi * 8];
      set8(af[mi], 0, pr);
      set8(af[mi], 8, pr + 16);
    }
#pragma unroll
    for (int ni = 0; ni < 4; ++ni) {
      // B 32x16 bf16 layout: lane col = l15; K chunks [hi*8,hi*8+8) and [hi*8+16,hi*8+24)
      const __bf16* pc = &Bt[cb][(wn + ni * 16 + l15) * LDB + hi * 8];
      set8(bfv[ni], 0, pc);
      set8(bfv[ni], 8, pc + 16);
    }
#pragma unroll
    for (int mi = 0; mi < 2; ++mi)
#pragma unroll
      for (int ni = 0; ni < 4; ++ni)
        acc[mi][ni] = __builtin_amdgcn_wmma_f32_16x16x32_bf16(
            false, af[mi], false, bfv[ni], (short)0, acc[mi][ni], false, false);
  }

  // C layout: VGPR r -> row = r + 8*hi, col = l15
#pragma unroll
  for (int mi = 0; mi < 2; ++mi)
#pragma unroll
    for (int ni = 0; ni < 4; ++ni) {
      const int col = bn + wn + ni * 16 + l15;
#pragma unroll
      for (int r = 0; r < 8; ++r) {
        const int row = bm + wm + mi * 16 + r + hi * 8;
        C[(size_t)row * N + col] = acc[mi][ni][r];
      }
    }
}

// ---------------------------------------------------------------------------
// RMSNorm + RoPE, fp32 in -> bf16 out, relayout [b][s][H*128] -> [b][H][s][128]
// ---------------------------------------------------------------------------
__global__ __launch_bounds__(128)
void normrope(const float* __restrict__ raw, const float* __restrict__ gsc,
              const float* __restrict__ cosb, const float* __restrict__ sinb,
              __bf16* __restrict__ outp, int H) {
  constexpr int S = 2048, HD = 128;
  const int spos = blockIdx.x, h = blockIdx.y, bb = blockIdx.z;
  const int t = threadIdx.x;
  const float* rowp = raw + ((size_t)bb * S + spos) * ((size_t)H * HD) + h * HD;
  const float v  = rowp[t];
  const float vp = rowp[t ^ 64];
  float ss = v * v;
#pragma unroll
  for (int m = 1; m < 32; m <<= 1) ss += __shfl_xor(ss, m, 32);
  __shared__ float part[4];
  if ((t & 31) == 0) part[t >> 5] = ss;
  __syncthreads();
  const float tot = part[0] + part[1] + part[2] + part[3];
  const float inv = rsqrtf(tot * (1.0f / HD) + 1e-6f);
  const float xn  = v  * inv * gsc[t];
  const float xpn = vp * inv * gsc[t ^ 64];
  const float rot = (t < 64) ? -xpn : xpn;
  const float res = xn * cosb[spos * HD + t] + rot * sinb[spos * HD + t];
  outp[(((size_t)bb * H + h) * S + spos) * HD + t] = (__bf16)res;
}

// ---------------------------------------------------------------------------
// V: fp32 [b][s][kv*128] -> bf16 [b][kv][s][128]
// ---------------------------------------------------------------------------
__global__ __launch_bounds__(128)
void reshape_cast_v(const float* __restrict__ vr, __bf16* __restrict__ vb) {
  constexpr int S = 2048, HD = 128, KVH = 4;
  const int spos = blockIdx.x, h = blockIdx.y, bb = blockIdx.z;
  const int t = threadIdx.x;
  vb[(((size_t)bb * KVH + h) * S + spos) * HD + t] =
      (__bf16)vr[((size_t)bb * S + spos) * (KVH * HD) + h * HD + t];
}

// ---------------------------------------------------------------------------
// Flash attention: one wave per (b, head, 16-row q tile). Causal, GQA (g=4).
// ---------------------------------------------------------------------------
__global__ __launch_bounds__(32)
void attn_fwd(const __bf16* __restrict__ Qn, const __bf16* __restrict__ Kn,
              const __bf16* __restrict__ Vb, __bf16* __restrict__ Ctx) {
  constexpr int S = 2048, HD = 128, NH = 16, KVH = 4;
  const int qt = blockIdx.x, h = blockIdx.y, bb = blockIdx.z;
  const int lane = threadIdx.x, hi = lane >> 4, l15 = lane & 15;

  const __bf16* Q  = Qn + (((size_t)(bb * NH + h)) * S + qt * 16) * HD;
  const __bf16* Kp = Kn + ((size_t)(bb * KVH + (h >> 2))) * S * HD;
  const __bf16* Vp = Vb + ((size_t)(bb * KVH + (h >> 2))) * S * HD;

  __shared__ __bf16 Ps[16 * 40];    // P tile, padded
  __shared__ __bf16 Vs[32 * 136];   // V tile [key][hd], padded stride (272B)

  // Q fragments: 4 k-steps over hd=128 (reused across all key blocks)
  v16bf qf[4];
#pragma unroll
  for (int ks = 0; ks < 4; ++ks) {
    const __bf16* qr = Q + (size_t)l15 * HD + ks * 32 + hi * 8;
    set8(qf[ks], 0, qr);
    set8(qf[ks], 8, qr + 16);
  }

  v8f ctx[8] = {};
  float mrow[8], lrow[8];
#pragma unroll
  for (int r = 0; r < 8; ++r) { mrow[r] = -__builtin_inff(); lrow[r] = 0.0f; }

  const float scale = 0.08838834764831845f;  // 1/sqrt(128)
  const int kmax = qt * 16 + 15;
  const int nkb  = (kmax >> 5) + 1;

  for (int kb = 0; kb < nkb; ++kb) {
    const int kbase = kb * 32;
    { // stage V tile: each lane copies one key row (256B) into LDS
      const __bf16* vr = Vp + (size_t)(kbase + lane) * HD;
#pragma unroll
      for (int c = 0; c < 16; ++c)
        *reinterpret_cast<BF8*>(&Vs[lane * 136 + c * 8]) =
            *reinterpret_cast<const BF8*>(vr + c * 8);
    }

    // S = Q * K^T : B fragments read contiguously from row-major K
    v8f sacc[2] = {};
#pragma unroll
    for (int ks = 0; ks < 4; ++ks) {
#pragma unroll
      for (int ni = 0; ni < 2; ++ni) {
        const int key = kbase + ni * 16 + l15;
        const __bf16* kr = Kp + (size_t)key * HD + ks * 32 + hi * 8;
        v16bf kf;
        set8(kf, 0, kr);
        set8(kf, 8, kr + 16);
        sacc[ni] = __builtin_amdgcn_wmma_f32_16x16x32_bf16(
            false, qf[ks], false, kf, (short)0, sacc[ni], false, false);
      }
    }

    // online softmax over 32 key columns (reduce across the 16-lane half)
#pragma unroll
    for (int r = 0; r < 8; ++r) {
      const int row = qt * 16 + r + hi * 8;
      float v0 = sacc[0][r] * scale;
      float v1 = sacc[1][r] * scale;
      if (kbase + l15 > row)      v0 = -__builtin_inff();
      if (kbase + 16 + l15 > row) v1 = -__builtin_inff();
      float mx = fmaxf(v0, v1);
#pragma unroll
      for (int m = 1; m < 16; m <<= 1) mx = fmaxf(mx, __shfl_xor(mx, m, 32));
      const float mnew = fmaxf(mrow[r], mx);
      const float e0 = __expf(v0 - mnew);
      const float e1 = __expf(v1 - mnew);
      float ps = e0 + e1;
#pragma unroll
      for (int m = 1; m < 16; m <<= 1) ps += __shfl_xor(ps, m, 32);
      const float corr = __expf(mrow[r] - mnew);
      lrow[r] = lrow[r] * corr + ps;
      mrow[r] = mnew;
#pragma unroll
      for (int ni = 0; ni < 8; ++ni) ctx[ni][r] *= corr;
      Ps[(r + hi * 8) * 40 + l15]      = (__bf16)e0;
      Ps[(r + hi * 8) * 40 + 16 + l15] = (__bf16)e1;
    }
    __syncthreads();

    // ctx += P * V   (P re-read in A-fragment layout; V fragments strided from LDS)
    v16bf pf;
    {
      const __bf16* pr = &Ps[l15 * 40 + hi * 16];
      set8(pf, 0, pr);
      set8(pf, 8, pr + 8);
    }
#pragma unroll
    for (int ni = 0; ni < 8; ++ni) {
      v16bf vf;
#pragma unroll
      for (int e = 0; e < 16; ++e)
        vf[e] = Vs[(e + hi * 16) * 136 + ni * 16 + l15];
      ctx[ni] = __builtin_amdgcn_wmma_f32_16x16x32_bf16(
          false, pf, false, vf, (short)0, ctx[ni], false, false);
    }
    __syncthreads();
  }

  // write ctx (bf16) in [b*s][nh*hd] layout for the output projection GEMM
#pragma unroll
  for (int ni = 0; ni < 8; ++ni) {
#pragma unroll
    for (int r = 0; r < 8; ++r) {
      const int rowl = r + hi * 8;
      const float inv = 1.0f / lrow[r];
      Ctx[((size_t)(bb * S + qt * 16 + rowl)) * (NH * HD) + h * HD + ni * 16 + l15] =
          (__bf16)(ctx[ni][r] * inv);
    }
  }
}

// ---------------------------------------------------------------------------
extern "C" void kernel_launch(void* const* d_in, const int* in_sizes, int n_in,
                              void* d_out, int out_size, void* d_ws, size_t ws_size,
                              hipStream_t stream) {
  const float* x    = (const float*)d_in[0];
  // d_in[1] = mask (unused; causal mask computed analytically)
  const float* cosb = (const float*)d_in[2];
  const float* sinb = (const float*)d_in[3];
  const float* Wq   = (const float*)d_in[4];
  const float* Wk   = (const float*)d_in[5];
  const float* Wv   = (const float*)d_in[6];
  const float* Wo   = (const float*)d_in[7];
  const float* qsc  = (const float*)d_in[8];
  const float* ksc  = (const float*)d_in[9];
  float* out = (float*)d_out;

  constexpr int B = 2, S = 2048, DIN = 2048, NH = 16, KVH = 4, HD = 128;
  constexpr int DOUT = NH * HD;   // 2048
  constexpr int KVD  = KVH * HD;  // 512
  constexpr long long MB = (long long)B * S;  // 4096

  char* base = (char*)d_ws;
  size_t off = 0;
  auto alloc = [&](size_t bytes) {
    char* p = base + off;
    off = (off + bytes + 255) & ~(size_t)255;
    return p;
  };
  __bf16* xb   = (__bf16*)alloc((size_t)MB * DIN * 2);
  __bf16* Wqb  = (__bf16*)alloc((size_t)DIN * DOUT * 2);
  __bf16* Wkb  = (__bf16*)alloc((size_t)DIN * KVD * 2);
  __bf16* Wvb  = (__bf16*)alloc((size_t)DIN * KVD * 2);
  __bf16* Wob  = (__bf16*)alloc((size_t)DOUT * DIN * 2);
  float*  qraw = (float*)alloc((size_t)MB * DOUT * 4);
  float*  kraw = (float*)alloc((size_t)MB * KVD * 4);
  float*  vraw = (float*)alloc((size_t)MB * KVD * 4);
  __bf16* qn   = (__bf16*)alloc((size_t)MB * DOUT * 2);
  __bf16* kn   = (__bf16*)alloc((size_t)MB * KVD * 2);
  __bf16* vb   = (__bf16*)alloc((size_t)MB * KVD * 2);
  __bf16* ctx  = (__bf16*)alloc((size_t)MB * DOUT * 2);

  cvt_f32_bf16<<<2048, 256, 0, stream>>>(x,  xb,  MB * DIN);
  cvt_f32_bf16<<<2048, 256, 0, stream>>>(Wq, Wqb, (long long)DIN * DOUT);
  cvt_f32_bf16<<<1024, 256, 0, stream>>>(Wk, Wkb, (long long)DIN * KVD);
  cvt_f32_bf16<<<1024, 256, 0, stream>>>(Wv, Wvb, (long long)DIN * KVD);
  cvt_f32_bf16<<<2048, 256, 0, stream>>>(Wo, Wob, (long long)DOUT * DIN);

  gemm_bf16<<<dim3(DOUT / GTN, MB / GTM), 256, 0, stream>>>(xb, Wqb, qraw, (int)MB, DOUT, DIN);
  gemm_bf16<<<dim3(KVD  / GTN, MB / GTM), 256, 0, stream>>>(xb, Wkb, kraw, (int)MB, KVD, DIN);
  gemm_bf16<<<dim3(KVD  / GTN, MB / GTM), 256, 0, stream>>>(xb, Wvb, vraw, (int)MB, KVD, DIN);

  normrope<<<dim3(S, NH,  B), 128, 0, stream>>>(qraw, qsc, cosb, sinb, qn, NH);
  normrope<<<dim3(S, KVH, B), 128, 0, stream>>>(kraw, ksc, cosb, sinb, kn, KVH);
  reshape_cast_v<<<dim3(S, KVH, B), 128, 0, stream>>>(vraw, vb);

  attn_fwd<<<dim3(S / 16, NH, B), 32, 0, stream>>>(qn, kn, vb, ctx);

  gemm_bf16<<<dim3(DIN / GTN, MB / GTM), 256, 0, stream>>>(ctx, Wob, out, (int)MB, DIN, DOUT);

  (void)in_sizes; (void)n_in; (void)out_size; (void)ws_size;
}
